// FakeBlock_90177133346911
// MI455X (gfx1250) — matrix operand order repaired and checked by reading, not compiled, not run on hardware
//
#include <hip/hip_runtime.h>
#include <math.h>

// ---------------------------------------------------------------------------
// Types for CDNA5 WMMA (wave32): 16x16x32 bf16 -> f32 accumulate
// ---------------------------------------------------------------------------
typedef __attribute__((ext_vector_type(16))) __bf16       v16bf;
typedef __attribute__((ext_vector_type(8)))  float        v8f;
typedef __attribute__((ext_vector_type(4)))  unsigned int u32x4;  // trivial, union-safe
typedef __attribute__((ext_vector_type(4)))  unsigned int ui32x4;
typedef __attribute__((ext_vector_type(8)))  int          i32x8;
typedef __attribute__((ext_vector_type(4)))  int          i32x4;

union Frag { v16bf v; u32x4 u[2]; };

__device__ __forceinline__ __bf16 f2bf(float f) { return (__bf16)f; }

__device__ __forceinline__ float gelu_exact(float v) {
  return 0.5f * v * (1.0f + erff(v * 0.70710678118654752f));
}

// ---------------------------------------------------------------------------
// Tensor Data Mover: issue a 2D bf16 tile load (tile_w x tile_h elements) from
// a row-major tensor (row stride = row_stride elements) into LDS at lds_addr.
// Descriptor bit-packing per CDNA5 ISA §8.3 (group 0) / §8.4 (group 1):
//   g0: [1:0]=count=1, [63:32]=lds_addr, [120:64]=global byte addr, [127:126]=2
//   g1: [17:16]=data_size(1=2B), [79:48]=tensor_dim0, [111:80]=tensor_dim1,
//       [127:112]=tile_dim0, [143:128]=tile_dim1, [207:160]=tensor_dim0_stride
// Groups 2/3 (and the trailing group) zero => plain 2D tensor.
// Tracked by TENSORcnt; wave waits with s_wait_tensorcnt 0.
// This toolchain exposes the 6-arg builtin:
//   (uint32x4 g0, int32x8 g1, int32x4, int32x4, int32x8, i32 cpol)
// ---------------------------------------------------------------------------
#if __has_builtin(__builtin_amdgcn_tensor_load_to_lds)
#define HAVE_TDM 1
__device__ __forceinline__ void tdm_load_2d_bf16(
    unsigned lds_addr, const void* gaddr,
    unsigned tile_w, unsigned tile_h,
    unsigned tensor_w, unsigned tensor_h, unsigned row_stride)
{
  const unsigned long long ga = (unsigned long long)gaddr;
  ui32x4 g0;
  g0[0] = 1u;                                               // count=1 (user)
  g0[1] = lds_addr;                                         // LDS byte addr
  g0[2] = (unsigned)(ga & 0xffffffffu);                     // addr[31:0]
  g0[3] = (unsigned)((ga >> 32) & 0x01ffffffu) | (2u << 30);// addr[56:32]|type=2
  i32x8 g1;
  g1[0] = (int)(1u << 16);                                  // data_size=2 bytes
  g1[1] = (int)((tensor_w & 0xffffu) << 16);                // dim0[15:0]
  g1[2] = (int)(((tensor_w >> 16) & 0xffffu) |
                ((tensor_h & 0xffffu) << 16));              // dim0[31:16]|dim1[15:0]
  g1[3] = (int)(((tensor_h >> 16) & 0xffffu) |
                ((tile_w & 0xffffu) << 16));                // dim1[31:16]|tile0
  g1[4] = (int)(tile_h & 0xffffu);                          // tile1 (tile2=0)
  g1[5] = (int)row_stride;                                  // dim0_stride[31:0]
  g1[6] = 0;                                                // stride hi / dim1_stride lo
  g1[7] = 0;
  const i32x4 z4 = {0, 0, 0, 0};
  const i32x8 z8 = {0, 0, 0, 0, 0, 0, 0, 0};
  __builtin_amdgcn_tensor_load_to_lds(g0, g1, z4, z4, z8, 0);
}
#else
#define HAVE_TDM 0
#endif

// ---------------------------------------------------------------------------
// fp32 [K][N] weights -> bf16 [N][K] (transposed) so GEMM B-tiles are
// K-contiguous per output column (matches WMMA B-fragment per-lane layout).
// ---------------------------------------------------------------------------
__global__ __launch_bounds__(256) void convert_transpose_w(
    const float* __restrict__ w, __bf16* __restrict__ wt, int K, int Nc)
{
  const size_t t = (size_t)blockIdx.x * 256 + threadIdx.x;
  if (t >= (size_t)K * Nc) return;
  const int k = (int)(t / Nc);
  const int n = (int)(t % Nc);
  wt[(size_t)n * K + k] = f2bf(w[t]);
}

// ---------------------------------------------------------------------------
// LayerNorm fp32 row -> bf16 row (one block per row)
// ---------------------------------------------------------------------------
__global__ __launch_bounds__(256) void layernorm_to_bf16(
    const float* __restrict__ x, const float* __restrict__ w,
    const float* __restrict__ b, __bf16* __restrict__ out, int C)
{
  const int row = blockIdx.x;
  const int t = threadIdx.x;
  const float* xr = x + (size_t)row * C;
  float s = 0.f, s2 = 0.f;
  for (int c = t; c < C; c += 256) { float v = xr[c]; s += v; s2 += v * v; }
  __shared__ float rs[256], rq[256];
  rs[t] = s; rq[t] = s2;
  __syncthreads();
  for (int off = 128; off > 0; off >>= 1) {
    if (t < off) { rs[t] += rs[t + off]; rq[t] += rq[t + off]; }
    __syncthreads();
  }
  const float mu   = rs[0] / C;
  const float var  = rq[0] / C - mu * mu;
  const float rinv = rsqrtf(var + 1e-5f);
  __bf16* orow = out + (size_t)row * C;
  for (int c = t; c < C; c += 256)
    orow[c] = f2bf((xr[c] - mu) * rinv * w[c] + b[c]);
}

// ---------------------------------------------------------------------------
// bf16 WMMA GEMM: out = A[M][K] * Bt[N][K]^T + bias, with epilogue variants.
// Block: 256 threads = 8 waves; 128x64 block tile; each wave does a 32x32
// patch = 2x2 v_wmma_f32_16x16x32_bf16 tiles; K-step 32.
// Tile staging via the Tensor Data Mover (one wave issues 2 descriptors per
// K-step; everyone else rides s_wait_tensorcnt + barrier).
//   EPI 0: out bf16 = acc + bias                 (QKV)
//   EPI 1: out bf16 = gelu(acc + bias)           (MLP up)
//   EPI 2: out f32  = acc + bias + resid         (proj / MLP down + residual)
// ---------------------------------------------------------------------------
template <int EPI>
__global__ __launch_bounds__(256) void gemm_bf16_wmma(
    const __bf16* __restrict__ A,    // [Mdim][Kdim] row-major bf16
    const __bf16* __restrict__ Bt,   // [Ndim][Kdim] row-major bf16 (B^T)
    const float*  __restrict__ bias, // [Ndim]
    const float*  __restrict__ resid,
    float*        __restrict__ outF,
    __bf16*       __restrict__ outB,
    int Mdim, int Ndim, int Kdim)
{
  __shared__ __bf16 As[128][32];
  __shared__ __bf16 Bs[64][32];

  const int t  = threadIdx.x;
  const int l  = t & 31;
  const int w  = t >> 5;
  const int wr = w >> 1;   // 0..3 : 32-row band
  const int wc = w & 1;    // 0..1 : 32-col band
  const int lm = l & 15;
  const int hi = l >> 4;
  const int m0 = blockIdx.y * 128;
  const int n0 = blockIdx.x * 64;

  v8f acc[2][2] = {};

#if HAVE_TDM
  const unsigned lds_as = (unsigned)(uintptr_t)&As[0][0];
  const unsigned lds_bs = (unsigned)(uintptr_t)&Bs[0][0];
#endif

  for (int k0 = 0; k0 < Kdim; k0 += 32) {
#if HAVE_TDM
    // Tensor DMA: A tile 128x32, B tile 64x32, both row-major into LDS.
    if (t < 32) {
      tdm_load_2d_bf16(lds_as, A + (size_t)m0 * Kdim + k0,
                       32u, 128u, (unsigned)Kdim, (unsigned)Mdim, (unsigned)Kdim);
      tdm_load_2d_bf16(lds_bs, Bt + (size_t)n0 * Kdim + k0,
                       32u, 64u, (unsigned)Kdim, (unsigned)Ndim, (unsigned)Kdim);
      __builtin_amdgcn_s_wait_tensorcnt(0);
    }
#else
    // Fallback: stage tiles through VGPRs.
#pragma unroll
    for (int i = 0; i < 2; ++i) {
      int e   = t + i * 256;
      int row = e >> 2;
      int c8  = (e & 3) * 8;
      *(u32x4*)&As[row][c8] =
          *(const u32x4*)(A + (size_t)(m0 + row) * Kdim + k0 + c8);
    }
    {
      int row = t >> 2;
      int c8  = (t & 3) * 8;
      *(u32x4*)&Bs[row][c8] =
          *(const u32x4*)(Bt + (size_t)(n0 + row) * Kdim + k0 + c8);
    }
    if (k0 + 32 < Kdim)
      __builtin_prefetch(A + (size_t)(m0 + (t >> 1)) * Kdim + k0 + 32, 0, 1);
#endif
    __syncthreads();

    // Fragment loads per ISA 7.12.2:
    //  A 16x32: lane M=l&15; dwords0-3 K=hi*8..+7, dwords4-7 K=16+hi*8..+7
    //  B 32x16: lane n=l&15; dwords0-3 K=hi*8..+7, dwords4-7 K=16+hi*8..+7
    Frag af[2], bfr[2];
#pragma unroll
    for (int tr = 0; tr < 2; ++tr) {
      const __bf16* p = &As[wr * 32 + tr * 16 + lm][0];
      af[tr].u[0] = *(const u32x4*)(p + hi * 8);
      af[tr].u[1] = *(const u32x4*)(p + 16 + hi * 8);
    }
#pragma unroll
    for (int tc = 0; tc < 2; ++tc) {
      const __bf16* p = &Bs[wc * 32 + tc * 16 + lm][0];
      bfr[tc].u[0] = *(const u32x4*)(p + hi * 8);
      bfr[tc].u[1] = *(const u32x4*)(p + 16 + hi * 8);
    }
#pragma unroll
    for (int tr = 0; tr < 2; ++tr)
#pragma unroll
      for (int tc = 0; tc < 2; ++tc)
        acc[tr][tc] = __builtin_amdgcn_wmma_f32_16x16x32_bf16(
            false, af[tr].v, false, bfr[tc].v, (short)0, acc[tr][tc],
            false, false);
    __syncthreads();
  }

  // epilogue; C/D layout: VGPR r, lanes0-15 -> M=r, lanes16-31 -> M=8+r; N=l&15
#pragma unroll
  for (int tr = 0; tr < 2; ++tr) {
#pragma unroll
    for (int tc = 0; tc < 2; ++tc) {
      const int   col = n0 + wc * 32 + tc * 16 + lm;
      const float bv  = bias[col];
#pragma unroll
      for (int r = 0; r < 8; ++r) {
        const int    row = m0 + wr * 32 + tr * 16 + r + 8 * hi;
        const size_t idx = (size_t)row * Ndim + col;
        const float  v   = acc[tr][tc][r] + bv;
        if (EPI == 0) outB[idx] = f2bf(v);
        if (EPI == 1) outB[idx] = f2bf(gelu_exact(v));
        if (EPI == 2) outF[idx] = v + resid[idx];
      }
    }
  }
}

// ---------------------------------------------------------------------------
// Flash attention, h=2, d=512, bf16 WMMA. One block per (b, h, 16-query tile).
// 4 waves; wave w owns output d-columns [w*128, w*128+128).
//  - Q fragments register-resident (A-layout matches row-major q rows).
//  - S = Q K^T: B-fragments loaded directly from global k rows (row-major k is
//    K-contiguous per key column); cross-wave d-slice reduction via LDS
//    ds_add_f32 atomics.
//  - online softmax (running max / sum / alpha per query row).
//  - O += P @ V with V transposed into padded LDS (aligned ds_load_b128 frags).
// ---------------------------------------------------------------------------
__global__ __launch_bounds__(128) void flash_attn_wmma(
    const __bf16* __restrict__ qkv,  // [B*N][3072]: q @ h*512, k @ 1024+h*512, v @ 2048+h*512
    __bf16* __restrict__ out,        // [B*N][1024]
    int N)
{
  const int nqt = N / 16;
  const int bid = blockIdx.x;
  const int q0  = (bid % nqt) * 16;
  const int h   = (bid / nqt) & 1;
  const int b   = bid / (nqt * 2);
  const int l   = threadIdx.x & 31;
  const int w   = threadIdx.x >> 5;   // wave 0..3
  const int lm  = l & 15;
  const int hi  = l >> 4;

  const size_t rs = 3072;
  const __bf16* qb = qkv + (size_t)b * N * rs + h * 512;
  const __bf16* kb = qb + 1024;
  const __bf16* vb = qb + 2048;

  __shared__ float  S[16][32];
  __shared__ __bf16 P[16][32];
  __shared__ __bf16 Vt[4][128][40];   // [wave][d-col][key]; 40 keeps rows 16B-aligned
  __shared__ float  mrun[16], lrun[16], alf[16];

  // Q fragments: 16 rows x this wave's 128-wide d-slice, 4 K-steps of 32
  v16bf qf[4];
  {
    const __bf16* qr = qb + (size_t)(q0 + lm) * rs + w * 128;
#pragma unroll
    for (int kk = 0; kk < 4; ++kk) {
      Frag f;
      f.u[0] = *(const u32x4*)(qr + kk * 32 + hi * 8);
      f.u[1] = *(const u32x4*)(qr + kk * 32 + 16 + hi * 8);
      qf[kk] = f.v;
    }
  }

  v8f o[8] = {};
  if (threadIdx.x < 16) { mrun[threadIdx.x] = -3.0e38f; lrun[threadIdx.x] = 0.0f; }

  const float scale = 0.044194173824159216f;  // 512^-0.5

  for (int kt = 0; kt < N / 32; ++kt) {
    for (int i = threadIdx.x; i < 16 * 32; i += 128) ((float*)S)[i] = 0.0f;
    __syncthreads();

    // S partials for this wave's d-slice, reduced into S via LDS float atomics
#pragma unroll
    for (int ct = 0; ct < 2; ++ct) {
      v8f s = {};
      const int key = kt * 32 + ct * 16 + lm;
      const __bf16* kr = kb + (size_t)key * rs + w * 128;
#pragma unroll
      for (int kk = 0; kk < 4; ++kk) {
        Frag f;
        f.u[0] = *(const u32x4*)(kr + kk * 32 + hi * 8);
        f.u[1] = *(const u32x4*)(kr + kk * 32 + 16 + hi * 8);
        s = __builtin_amdgcn_wmma_f32_16x16x32_bf16(false, qf[kk], false, f.v,
                                                    (short)0, s, false, false);
      }
#pragma unroll
      for (int r = 0; r < 8; ++r)
        atomicAdd(&S[r + 8 * hi][ct * 16 + lm], s[r]);
    }

    // stage V^T for this wave's d-slice (lane <-> key)
    {
      const __bf16* vr = vb + (size_t)(kt * 32 + l) * rs + w * 128;
#pragma unroll
      for (int i = 0; i < 16; ++i) {
        u32x4 d = *(const u32x4*)(vr + i * 8);
        const __bf16* e = (const __bf16*)&d;
#pragma unroll
        for (int j = 0; j < 8; ++j) Vt[w][i * 8 + j][l] = e[j];
      }
    }
    __syncthreads();

    // online softmax over this 32-key tile (one thread per query row)
    if (threadIdx.x < 16) {
      const int r = threadIdx.x;
      float tmax = -3.0e38f;
      for (int j = 0; j < 32; ++j) tmax = fmaxf(tmax, S[r][j] * scale);
      const float mo = mrun[r];
      const float mn = fmaxf(mo, tmax);
      const float a  = __expf(mo - mn);
      float ls = 0.0f;
      for (int j = 0; j < 32; ++j) {
        const float p = __expf(S[r][j] * scale - mn);
        ls += p;
        P[r][j] = f2bf(p);
      }
      lrun[r] = lrun[r] * a + ls;
      mrun[r] = mn;
      alf[r]  = a;
    }
    __syncthreads();

    // O = O * alpha + P @ V
    float av[8];
#pragma unroll
    for (int r = 0; r < 8; ++r) av[r] = alf[r + 8 * hi];
    Frag pf;
    pf.u[0] = *(const u32x4*)&P[lm][hi * 8];
    pf.u[1] = *(const u32x4*)&P[lm][16 + hi * 8];
#pragma unroll
    for (int ct = 0; ct < 8; ++ct) {
#pragma unroll
      for (int r = 0; r < 8; ++r) o[ct][r] = o[ct][r] * av[r];
      Frag vf;
      const __bf16* p = &Vt[w][ct * 16 + lm][0];
      vf.u[0] = *(const u32x4*)(p + hi * 8);
      vf.u[1] = *(const u32x4*)(p + 16 + hi * 8);
      o[ct] = __builtin_amdgcn_wmma_f32_16x16x32_bf16(false, pf.v, false, vf.v,
                                                      (short)0, o[ct], false, false);
    }
    __syncthreads();
  }

  // normalize by running sum and store bf16
  float inv[8];
#pragma unroll
  for (int r = 0; r < 8; ++r) inv[r] = 1.0f / lrun[r + 8 * hi];
#pragma unroll
  for (int ct = 0; ct < 8; ++ct) {
#pragma unroll
    for (int r = 0; r < 8; ++r) {
      const int row = q0 + r + 8 * hi;
      const int col = h * 512 + w * 128 + ct * 16 + lm;
      out[((size_t)b * N + row) * 1024 + col] = f2bf(o[ct][r] * inv[r]);
    }
  }
}

// ---------------------------------------------------------------------------
// Host orchestration
// ---------------------------------------------------------------------------
extern "C" void kernel_launch(void* const* d_in, const int* in_sizes, int n_in,
                              void* d_out, int out_size, void* d_ws, size_t ws_size,
                              hipStream_t stream) {
  const int B = 8, N = 2048, C = 1024, H = 2048;
  const int M = B * N;  // 16384 rows
  const float* x      = (const float*)d_in[0];
  const float* ln1_w  = (const float*)d_in[1];
  const float* ln1_b  = (const float*)d_in[2];
  const float* qkv_w  = (const float*)d_in[3];
  const float* qkv_b  = (const float*)d_in[4];
  const float* proj_w = (const float*)d_in[5];
  const float* proj_b = (const float*)d_in[6];
  const float* ln2_w  = (const float*)d_in[7];
  const float* ln2_b  = (const float*)d_in[8];
  const float* mlp_w1 = (const float*)d_in[9];
  const float* mlp_b1 = (const float*)d_in[10];
  const float* mlp_w2 = (const float*)d_in[11];
  const float* mlp_b2 = (const float*)d_in[12];
  float* outp = (float*)d_out;
  (void)in_sizes; (void)n_in; (void)out_size; (void)ws_size;

  char* ws = (char*)d_ws;
  size_t off = 0;
  auto take = [&](size_t bytes) -> void* {
    off = (off + 255) & ~(size_t)255;
    void* p = ws + off;
    off += bytes;
    return p;
  };
  __bf16* xn1     = (__bf16*)take((size_t)M * C * 2);
  __bf16* qkv_wt  = (__bf16*)take((size_t)3 * C * C * 2);
  __bf16* qkvo    = (__bf16*)take((size_t)M * 3 * C * 2);
  __bf16* attn_o  = (__bf16*)take((size_t)M * C * 2);
  __bf16* proj_wt = (__bf16*)take((size_t)C * C * 2);
  float*  x1      = (float*)take((size_t)M * C * 4);
  __bf16* xn2     = (__bf16*)take((size_t)M * C * 2);
  __bf16* w1t     = (__bf16*)take((size_t)C * H * 2);
  __bf16* hbuf    = (__bf16*)take((size_t)M * H * 2);
  __bf16* w2t     = (__bf16*)take((size_t)H * C * 2);

  // one-time weight convert + transpose (fp32 -> bf16 [N][K])
  convert_transpose_w<<<(3 * C * C + 255) / 256, 256, 0, stream>>>(qkv_w, qkv_wt, C, 3 * C);
  convert_transpose_w<<<(C * C + 255) / 256, 256, 0, stream>>>(proj_w, proj_wt, C, C);
  convert_transpose_w<<<(C * H + 255) / 256, 256, 0, stream>>>(mlp_w1, w1t, C, H);
  convert_transpose_w<<<(H * C + 255) / 256, 256, 0, stream>>>(mlp_w2, w2t, H, C);

  // attention branch
  layernorm_to_bf16<<<M, 256, 0, stream>>>(x, ln1_w, ln1_b, xn1, C);
  gemm_bf16_wmma<0><<<dim3(3 * C / 64, M / 128), 256, 0, stream>>>(
      xn1, qkv_wt, qkv_b, nullptr, nullptr, qkvo, M, 3 * C, C);
  flash_attn_wmma<<<B * 2 * (N / 16), 128, 0, stream>>>(qkvo, attn_o, N);
  gemm_bf16_wmma<2><<<dim3(C / 64, M / 128), 256, 0, stream>>>(
      attn_o, proj_wt, proj_b, x, x1, nullptr, M, C, C);

  // mlp branch
  layernorm_to_bf16<<<M, 256, 0, stream>>>(x1, ln2_w, ln2_b, xn2, C);
  gemm_bf16_wmma<1><<<dim3(H / 64, M / 128), 256, 0, stream>>>(
      xn2, w1t, mlp_b1, nullptr, nullptr, hbuf, M, H, C);
  gemm_bf16_wmma<2><<<dim3(C / 64, M / 128), 256, 0, stream>>>(
      hbuf, w2t, mlp_b2, x1, outp, nullptr, M, C, H);
}
